// MyViT_16879221473424
// MI455X (gfx1250) — matrix operand order repaired and checked by reading, not compiled, not run on hardware
//
#include <hip/hip_runtime.h>
#include <hip/hip_bf16.h>
#include <math.h>

typedef __attribute__((ext_vector_type(16))) _Float16 v16h;
typedef __attribute__((ext_vector_type(8)))  _Float16 v8h;
typedef __attribute__((ext_vector_type(8)))  float    v8f;

constexpr int Bn   = 64;
constexpr int Cn   = 9;
constexpr int IMHn = 128;
constexpr int NPn  = 8;
constexpr int PSn  = 16;
constexpr int Dn   = 1024;
constexpr int NHn  = 8;
constexpr int Ln   = 4;
constexpr int DHn  = 128;
constexpr int FFn  = 4096;
constexpr int INDn = 2304;      // C*PS*PS
constexpr int Sn   = 65;
constexpr int MT   = Bn * Sn;        // 4160 token rows (65 * 64)
constexpr int MP   = Bn * NPn * NPn; // 4096 patch rows  (64 * 64)

constexpr int BROW = 40;             // padded LDS row stride (halves) for B tile

// =====================================================================
// GEMM: C[M,N] = A[M,K](f16 row-major) * W[N,K](f16 row-major) + bias
// 4-wave block computes a 64x64 tile. The 64x32 B tile is staged in LDS
// via double-buffered per-lane global_load_async_to_lds_b128 (ASYNCcnt),
// shared by all 4 waves; A fragments stream from global (prefetched).
// Fragment layouts follow CDNA5 ISA 7.12.2 (wave32).
// =====================================================================
template<bool HALF_OUT, bool GELU>
__global__ __launch_bounds__(128)
void gemm_wmma(const _Float16* __restrict__ A, int lda, long aBatch,
               const _Float16* __restrict__ W, long wBatch,
               const float* __restrict__ bias, long biasBatch,
               void* __restrict__ Cout, int ldc, long cBatch,
               int M, int N, int K)
{
    __shared__ _Float16 Bs[2][64 * BROW];   // 2 x 5 KB double buffer

    const int z    = blockIdx.z;
    const int tid  = threadIdx.x;
    const int wave = tid >> 5;
    const int lane = tid & 31;
    const int r    = lane & 15;
    const int kh   = lane >> 4;     // K-half selector

    const _Float16* Ab = A + (long)z * aBatch +
                         (long)(blockIdx.x * 64 + wave * 16 + r) * lda;
    const _Float16* Wb = W + (long)z * wBatch;
    const int n0 = blockIdx.y * 64;

    // async B staging: each of 128 threads copies 32 contiguous bytes
    const int brow = tid >> 1;              // 0..63
    const int bcol = (tid & 1) * 16;        // halves within the 32-K slab
    const _Float16* gB = Wb + (long)(n0 + brow) * K + bcol;

    auto issueB = [&](int buf, int k0) {
        const _Float16* g = gB + k0;
        unsigned l = (unsigned)(size_t)&Bs[buf][brow * BROW + bcol];
        asm volatile("global_load_async_to_lds_b128 %0, %1, off"
                     :: "v"(l), "v"(g) : "memory");
        asm volatile("global_load_async_to_lds_b128 %0, %1, off"
                     :: "v"(l + 16u), "v"(g + 8) : "memory");
    };

    v8f zero = {};
    v8f acc[4] = {zero, zero, zero, zero};

    const int nk = K / 32;
    issueB(0, 0);
    for (int ki = 0; ki < nk; ++ki) {
        const int k0 = ki * 32;
        if (ki + 1 < nk) {
            issueB((ki + 1) & 1, k0 + 32);
            // 2 loads of the next buffer in flight; <=2 means current done
            asm volatile("s_wait_asynccnt 0x2" ::: "memory");
        } else {
            asm volatile("s_wait_asynccnt 0x0" ::: "memory");
        }
        __syncthreads();

        // A-fragment straight from global: halves 0..7 = K[k0+kh*8 .. +7],
        //                                  halves 8..15 = K[k0+16+kh*8 .. +7]
        union { v16h v; v8h p[2]; } a;
        const _Float16* pa = Ab + k0 + kh * 8;
        a.p[0] = *(const v8h*)(pa);
        a.p[1] = *(const v8h*)(pa + 16);
        __builtin_prefetch(pa + 256, 0, 1);   // global_prefetch_b8 ahead in K

        const _Float16* bbase = &Bs[ki & 1][0];
#pragma unroll
        for (int j = 0; j < 4; ++j) {
            // B-fragment: halves 0..7 = K[k0+kh*8 ..], halves 8..15 = K[k0+16+kh*8 ..] of column row
            const _Float16* pb = bbase + (j * 16 + r) * BROW + kh * 8;
            union { v16h v; v8h p[2]; } bf;
            bf.p[0] = *(const v8h*)(pb);
            bf.p[1] = *(const v8h*)(pb + 16);
            acc[j] = __builtin_amdgcn_wmma_f32_16x16x32_f16(
                false, a.v, false, bf.v, (short)0, acc[j], false, false);
        }
        __syncthreads();   // protect Bs[buf] before it is refilled
    }

    // C layout: VGPR v -> row = 16*tile + v + 8*kh, col = 16*ntile + r
#pragma unroll
    for (int j = 0; j < 4; ++j) {
        int col = n0 + j * 16 + r;
        if (col >= N) continue;
        float bval = bias ? bias[(long)z * biasBatch + col] : 0.0f;
#pragma unroll
        for (int v = 0; v < 8; ++v) {
            int row = blockIdx.x * 64 + wave * 16 + v + 8 * kh;
            if (row >= M) continue;
            float x = acc[j][v] + bval;
            if (GELU) x = 0.5f * x * (1.0f + erff(x * 0.70710678118654752f));
            long idx = (long)z * cBatch + (long)row * ldc + col;
            if (HALF_OUT) ((_Float16*)Cout)[idx] = (_Float16)x;
            else          ((float*)Cout)[idx]    = x;
        }
    }
}

// =====================================================================
// Fused attention: one wave per (b, h, s). Scores + softmax + P*V.
// Q/K/V layout: [NH][B*S][DH] f16.
// =====================================================================
__global__ __launch_bounds__(32)
void attention_kernel(const _Float16* __restrict__ Qh, const _Float16* __restrict__ Kh,
                      const _Float16* __restrict__ Vh, float* __restrict__ ctx)
{
    const int s    = blockIdx.x;
    const int bh   = blockIdx.y;
    const int bi   = bh / NHn;
    const int h    = bh - bi * NHn;
    const int lane = threadIdx.x;
    const long headOff = (long)h * MT * DHn;
    const long rowBase = (long)(bi * Sn) * DHn;
    const _Float16* q  = Qh + headOff + rowBase + (long)s * DHn;
    const _Float16* Kb = Kh + headOff + rowBase;
    const _Float16* Vb = Vh + headOff + rowBase;

    __shared__ float qs[DHn];
#pragma unroll
    for (int j = 0; j < 4; ++j) qs[lane + 32 * j] = (float)q[lane + 32 * j];
    __syncthreads();

    const float scale = 0.08838834764831845f;  // 1/sqrt(128)
    float sc[3];
#pragma unroll
    for (int i = 0; i < 3; ++i) {
        int t = lane + 32 * i;
        if (t < Sn) {
            const _Float16* kr = Kb + (long)t * DHn;
            float d = 0.0f;
#pragma unroll 8
            for (int e = 0; e < DHn; ++e) d += qs[e] * (float)kr[e];
            sc[i] = d * scale;
        } else {
            sc[i] = -1e30f;
        }
    }
    float mx = fmaxf(sc[0], fmaxf(sc[1], sc[2]));
#pragma unroll
    for (int off = 16; off > 0; off >>= 1) mx = fmaxf(mx, __shfl_xor(mx, off));

    float pr[3]; float sum = 0.0f;
#pragma unroll
    for (int i = 0; i < 3; ++i) {
        int t = lane + 32 * i;
        pr[i] = (t < Sn) ? __expf(sc[i] - mx) : 0.0f;
        sum += pr[i];
    }
#pragma unroll
    for (int off = 16; off > 0; off >>= 1) sum += __shfl_xor(sum, off);
    const float inv = 1.0f / sum;

    float o0 = 0, o1 = 0, o2 = 0, o3 = 0;
#pragma unroll
    for (int i = 0; i < 3; ++i) {
        const int base = 32 * i;
        int cnt = Sn - base; if (cnt > 32) cnt = 32;
        float pi = pr[i];
        for (int tt = 0; tt < cnt; ++tt) {
            float pt = __shfl(pi, tt);
            const _Float16* vr = Vb + (long)(base + tt) * DHn;
            o0 += pt * (float)vr[lane];
            o1 += pt * (float)vr[lane + 32];
            o2 += pt * (float)vr[lane + 64];
            o3 += pt * (float)vr[lane + 96];
        }
    }
    float* crow = ctx + (long)(bi * Sn + s) * Dn + h * DHn;
    crow[lane]      = o0 * inv;
    crow[lane + 32] = o1 * inv;
    crow[lane + 64] = o2 * inv;
    crow[lane + 96] = o3 * inv;
}

// =====================================================================
// LayerNorm over D=1024 per row; optional f16 output (GEMM feed).
// =====================================================================
template<bool HALF_OUT>
__global__ __launch_bounds__(256)
void layernorm_kernel(const float* __restrict__ Xin, const float* __restrict__ g,
                      const float* __restrict__ b, void* __restrict__ out)
{
    __shared__ float red[8];
    __shared__ float stat[2];
    const int row = blockIdx.x, tid = threadIdx.x;
    const float* x = Xin + (long)row * Dn;
    float v[4]; float s = 0.0f;
#pragma unroll
    for (int j = 0; j < 4; ++j) { v[j] = x[tid + 256 * j]; s += v[j]; }
#pragma unroll
    for (int off = 16; off > 0; off >>= 1) s += __shfl_xor(s, off);
    if ((tid & 31) == 0) red[tid >> 5] = s;
    __syncthreads();
    if (tid == 0) {
        float t = 0.0f;
        for (int i = 0; i < 8; ++i) t += red[i];
        stat[0] = t * (1.0f / Dn);
    }
    __syncthreads();
    const float mean = stat[0];
    float s2 = 0.0f;
#pragma unroll
    for (int j = 0; j < 4; ++j) { float d = v[j] - mean; s2 += d * d; }
#pragma unroll
    for (int off = 16; off > 0; off >>= 1) s2 += __shfl_xor(s2, off);
    if ((tid & 31) == 0) red[tid >> 5] = s2;
    __syncthreads();
    if (tid == 0) {
        float t = 0.0f;
        for (int i = 0; i < 8; ++i) t += red[i];
        stat[1] = rsqrtf(t * (1.0f / Dn) + 1e-5f);
    }
    __syncthreads();
    const float rs = stat[1];
#pragma unroll
    for (int j = 0; j < 4; ++j) {
        int col = tid + 256 * j;
        float y = (v[j] - mean) * rs * g[col] + b[col];
        if (HALF_OUT) ((_Float16*)out)[(long)row * Dn + col] = (_Float16)y;
        else          ((float*)out)[(long)row * Dn + col]    = y;
    }
}

// ============================ elementwise ============================
__global__ __launch_bounds__(256)
void patchify_kernel(const float* __restrict__ img, _Float16* __restrict__ A0)
{
    long idx = (long)blockIdx.x * 256 + threadIdx.x;
    if (idx >= (long)MP * INDn) return;
    int  i  = (int)(idx % INDn);
    long m  = idx / INDn;
    int  bi = (int)(m >> 6);
    int  pp = (int)(m & 63);
    int  py = pp >> 3, px = pp & 7;
    int  c  = i / 256;
    int  rr = i & 255;
    int  ph = rr >> 4, pw = rr & 15;
    int  y  = py * PSn + ph, x = px * PSn + pw;
    A0[idx] = (_Float16)img[(((long)bi * Cn + c) * IMHn + y) * IMHn + x];
}

// cls token + patch bias + sinusoidal positional embedding -> X
__global__ __launch_bounds__(256)
void assemble_kernel(const float* __restrict__ T, const float* __restrict__ b_map,
                     const float* __restrict__ cls, float* __restrict__ X)
{
    long idx = (long)blockIdx.x * 256 + threadIdx.x;
    if (idx >= (long)MT * Dn) return;
    int  d   = (int)(idx & (Dn - 1));
    long row = idx >> 10;
    int  bi  = (int)(row / Sn);
    int  s   = (int)(row - (long)bi * Sn);
    float freq = __powf(10000.0f, -(float)(d & ~1) * (1.0f / Dn));
    float ang  = (float)s * freq;
    float pos  = (d & 1) ? __cosf(ang) : __sinf(ang);
    float val  = (s == 0) ? cls[d]
                          : T[((long)(bi * 64 + (s - 1))) * Dn + d] + b_map[d];
    X[idx] = val + pos;
}

__global__ __launch_bounds__(256)
void cvt_f16_kernel(const float* __restrict__ src, _Float16* __restrict__ dst, long n)
{
    long i = (long)blockIdx.x * 256 + threadIdx.x;
    if (i < n) dst[i] = (_Float16)src[i];
}

__global__ __launch_bounds__(256)
void add_kernel(float* __restrict__ X, const float* __restrict__ R, long n)
{
    long i = (long)blockIdx.x * 256 + threadIdx.x;
    if (i < n) X[i] += R[i];
}

// =====================================================================
extern "C" void kernel_launch(void* const* d_in, const int* in_sizes, int n_in,
                              void* d_out, int out_size, void* d_ws, size_t ws_size,
                              hipStream_t stream)
{
    (void)in_sizes; (void)n_in; (void)out_size; (void)ws_size;
    const float* images = (const float*)d_in[0];
    const float* W_map  = (const float*)d_in[1];
    const float* b_map  = (const float*)d_in[2];
    const float* cls    = (const float*)d_in[3];
    const float* Wq     = (const float*)d_in[4];
    const float* bq     = (const float*)d_in[5];
    const float* Wk     = (const float*)d_in[6];
    const float* bk     = (const float*)d_in[7];
    const float* Wv     = (const float*)d_in[8];
    const float* bv     = (const float*)d_in[9];
    const float* ln1_g  = (const float*)d_in[10];
    const float* ln1_b  = (const float*)d_in[11];
    const float* ln2_g  = (const float*)d_in[12];
    const float* ln2_b  = (const float*)d_in[13];
    const float* W1     = (const float*)d_in[14];
    const float* b1     = (const float*)d_in[15];
    const float* W2     = (const float*)d_in[16];
    const float* b2     = (const float*)d_in[17];
    const float* lnf_g  = (const float*)d_in[18];
    const float* lnf_b  = (const float*)d_in[19];

    char* p = (char*)d_ws;
    auto take = [&](size_t bytes) {
        char* q = p; p += (bytes + 255) & ~(size_t)255; return q;
    };
    _Float16* Wmap16 = (_Float16*)take((size_t)Dn * INDn * 2);
    _Float16* Wq16   = (_Float16*)take((size_t)Ln * NHn * DHn * DHn * 2);
    _Float16* Wk16   = (_Float16*)take((size_t)Ln * NHn * DHn * DHn * 2);
    _Float16* Wv16   = (_Float16*)take((size_t)Ln * NHn * DHn * DHn * 2);
    _Float16* Wff1   = (_Float16*)take((size_t)FFn * Dn * 2);   // per-layer
    _Float16* Wff2   = (_Float16*)take((size_t)Dn * FFn * 2);   // per-layer
    _Float16* Hbuf   = (_Float16*)take((size_t)MT * FFn * 2);   // FF hidden; reused as A0
    float*    Tbuf   = (float*)take((size_t)MT * Dn * 4);       // embed out / ctx / FF2 out
    float*    X      = (float*)take((size_t)MT * Dn * 4);       // residual stream
    _Float16* Xn     = (_Float16*)take((size_t)MT * Dn * 2);    // LN output (GEMM A)
    _Float16* Q16    = (_Float16*)take((size_t)NHn * MT * DHn * 2);
    _Float16* K16    = (_Float16*)take((size_t)NHn * MT * DHn * 2);
    _Float16* V16    = (_Float16*)take((size_t)NHn * MT * DHn * 2);

    auto ew = [](long n) { return dim3((unsigned)((n + 255) / 256)); };

    // ---- weight conversions (small; QKV for all layers at once) ----
    cvt_f16_kernel<<<ew((long)Dn * INDn), 256, 0, stream>>>(W_map, Wmap16, (long)Dn * INDn);
    long qkvN = (long)Ln * NHn * DHn * DHn;
    cvt_f16_kernel<<<ew(qkvN), 256, 0, stream>>>(Wq, Wq16, qkvN);
    cvt_f16_kernel<<<ew(qkvN), 256, 0, stream>>>(Wk, Wk16, qkvN);
    cvt_f16_kernel<<<ew(qkvN), 256, 0, stream>>>(Wv, Wv16, qkvN);

    // ---- patchify + embedding GEMM [4096 x 1024 x 2304] ----
    _Float16* A0 = Hbuf;
    patchify_kernel<<<ew((long)MP * INDn), 256, 0, stream>>>(images, A0);
    gemm_wmma<false, false><<<dim3(MP / 64, Dn / 64, 1), 128, 0, stream>>>(
        A0, INDn, 0, Wmap16, 0, nullptr, 0, Tbuf, Dn, 0, MP, Dn, INDn);
    assemble_kernel<<<ew((long)MT * Dn), 256, 0, stream>>>(Tbuf, b_map, cls, X);

    for (int l = 0; l < Ln; ++l) {
        // ---- MSA ----
        layernorm_kernel<true><<<MT, 256, 0, stream>>>(X, ln1_g + l * Dn, ln1_b + l * Dn, Xn);
        const long wqkvB = (long)DHn * DHn;   // per-head weight stride
        const long cB    = (long)MT * DHn;    // per-head output stride
        gemm_wmma<true, false><<<dim3(MT / 64, DHn / 64, NHn), 128, 0, stream>>>(
            Xn, Dn, DHn, Wq16 + (size_t)l * NHn * DHn * DHn, wqkvB,
            bq + (size_t)l * NHn * DHn, DHn, Q16, DHn, cB, MT, DHn, DHn);
        gemm_wmma<true, false><<<dim3(MT / 64, DHn / 64, NHn), 128, 0, stream>>>(
            Xn, Dn, DHn, Wk16 + (size_t)l * NHn * DHn * DHn, wqkvB,
            bk + (size_t)l * NHn * DHn, DHn, K16, DHn, cB, MT, DHn, DHn);
        gemm_wmma<true, false><<<dim3(MT / 64, DHn / 64, NHn), 128, 0, stream>>>(
            Xn, Dn, DHn, Wv16 + (size_t)l * NHn * DHn * DHn, wqkvB,
            bv + (size_t)l * NHn * DHn, DHn, V16, DHn, cB, MT, DHn, DHn);
        attention_kernel<<<dim3(Sn, Bn * NHn), 32, 0, stream>>>(Q16, K16, V16, Tbuf);
        add_kernel<<<ew((long)MT * Dn), 256, 0, stream>>>(X, Tbuf, (long)MT * Dn);

        // ---- FFN ----
        layernorm_kernel<true><<<MT, 256, 0, stream>>>(X, ln2_g + l * Dn, ln2_b + l * Dn, Xn);
        cvt_f16_kernel<<<ew((long)FFn * Dn), 256, 0, stream>>>(W1 + (size_t)l * FFn * Dn, Wff1, (long)FFn * Dn);
        cvt_f16_kernel<<<ew((long)Dn * FFn), 256, 0, stream>>>(W2 + (size_t)l * Dn * FFn, Wff2, (long)Dn * FFn);
        gemm_wmma<true, true><<<dim3(MT / 64, FFn / 64, 1), 128, 0, stream>>>(
            Xn, Dn, 0, Wff1, 0, b1 + (size_t)l * FFn, 0, Hbuf, FFn, 0, MT, FFn, Dn);
        gemm_wmma<false, false><<<dim3(MT / 64, Dn / 64, 1), 128, 0, stream>>>(
            Hbuf, FFn, 0, Wff2, 0, b2 + (size_t)l * Dn, 0, Tbuf, Dn, 0, MT, Dn, FFn);
        add_kernel<<<ew((long)MT * Dn), 256, 0, stream>>>(X, Tbuf, (long)MT * Dn);
    }

    layernorm_kernel<false><<<MT, 256, 0, stream>>>(X, lnf_g, lnf_b, (float*)d_out);
}
